// Embed_59167469470004
// MI455X (gfx1250) — matrix operand order repaired and checked by reading, not compiled, not run on hardware
//
#include <hip/hip_runtime.h>
#include <hip/hip_bf16.h>

#define VOCAB 50257
#define DIM   1024
#define KSLICES 8           // K split: 8 slices of 128
#define KSLICE_LEN (DIM / KSLICES)

typedef __attribute__((ext_vector_type(16))) _Float16 v16h;
typedef __attribute__((ext_vector_type(8)))  float    v8f;
typedef __attribute__((ext_vector_type(4)))  float    v4f;

// ---------------------------------------------------------------------------
// Kernel 1: y[j] = (j == DIM-1) ? filters[j,0] : relu(filters[j,0])
// Writes f32 copy (for gate/finalize) and f16 copy (WMMA B operand).
// ---------------------------------------------------------------------------
__global__ __launch_bounds__(256) void prep_y_kernel(const float* __restrict__ filters,
                                                     float* __restrict__ y,
                                                     _Float16* __restrict__ yh) {
    const int j = blockIdx.x * blockDim.x + threadIdx.x;
    if (j < DIM) {
        const float f0 = filters[(size_t)j * VOCAB];   // column 0, row j
        const float yv = (j == DIM - 1) ? f0 : fmaxf(f0, 0.0f);
        y[j]  = yv;
        yh[j] = (_Float16)yv;
    }
}

// ---------------------------------------------------------------------------
// Kernel 2: partial GEMVs u = W_t @ y, v = W_h @ y via v_wmma_f32_16x16x32_f16.
// 128 row-tiles (64 for W_t, 64 for W_h) x 8 K-slices = 1024 waves, so ~4 MB
// of loads are in flight chip-wide (vs 0.5 MB with unsplit K) -> HBM-latency
// no longer caps bandwidth on the 8 MB weight stream.
//
// A layout: lane half h holds K = k0 + 16h + e (elems 0..15), matching B.
// B layout (contiguous order, same as A): lane half h holds K = k0 + 16h + e
//   -> one aligned 32-byte v16h load from the f16 y copy (broadcast over N).
// D layout: VGPR r -> M=r (lanes 0-15) / M=r+8 (lanes 16-31); all N equal.
//
// Each wave writes 16 partial dot products to part[slice][row2048]; every slot
// is written exactly once (no atomics, no zero-init, deterministic).
// ---------------------------------------------------------------------------
__global__ __launch_bounds__(256) void gemv_wmma_kernel(const float* __restrict__ Wt,
                                                        const float* __restrict__ Wh,
                                                        const _Float16* __restrict__ yh,
                                                        float* __restrict__ part) {
    const int lane = threadIdx.x & 31;
    const int wave = threadIdx.x >> 5;
    const int gw   = blockIdx.x * 8 + wave;          // 0..1023
    const int tile  = gw >> 3;                       // 0..127
    const int slice = gw & (KSLICES - 1);            // 0..7
    const bool isT = (tile < 64);
    const int  rowTile = isT ? tile : (tile - 64);   // 0..63
    const float* __restrict__ W = isT ? Wt : Wh;

    const int row  = rowTile * 16 + (lane & 15);     // M for this lane
    const int half = lane >> 4;                      // lane-half (K grouping)
    const int kbeg = slice * KSLICE_LEN;

    v8f acc = {};

    #pragma unroll
    for (int kk = 0; kk < KSLICE_LEN; kk += 32) {
        const int k0 = kbeg + kk;
        // ---- A tile: 16 f32 -> f16, per documented A layout ----
        const float* wp = W + (size_t)row * DIM + k0 + half * 16;
        const v4f a0 = *(const v4f*)(wp);
        const v4f a1 = *(const v4f*)(wp + 4);
        const v4f a2 = *(const v4f*)(wp + 8);
        const v4f a3 = *(const v4f*)(wp + 12);
        v16h A;
        #pragma unroll
        for (int i = 0; i < 4; ++i) {
            A[i]      = (_Float16)a0[i];
            A[4 + i]  = (_Float16)a1[i];
            A[8 + i]  = (_Float16)a2[i];
            A[12 + i] = (_Float16)a3[i];
        }
        // ---- B: 16 contiguous f16 y values, broadcast across all N ----
        const v16h B = *(const v16h*)(yh + k0 + half * 16);

        acc = __builtin_amdgcn_wmma_f32_16x16x32_f16(
            /*neg_a=*/false, A, /*neg_b=*/false, B,
            /*c_mod=*/(short)0, acc, /*reuse_a=*/false, /*reuse_b=*/false);
    }

    if ((lane & 15) == 0) {
        const int base = (isT ? 0 : DIM) + rowTile * 16 + half * 8;  // 0..2047
        float* dst = part + slice * (2 * DIM) + base;
        #pragma unroll
        for (int r = 0; r < 8; ++r) dst[r] = acc[r];
    }
}

// ---------------------------------------------------------------------------
// Kernel 3: fold K-slice partials, gate, log_softmax.
// Single block of 1024 threads (32 wave32 waves), LDS tree reductions.
// ---------------------------------------------------------------------------
__global__ __launch_bounds__(1024) void finalize_kernel(const float* __restrict__ y,
                                                        const float* __restrict__ part,
                                                        float* __restrict__ out) {
    __shared__ float sred[32];
    __shared__ float sbc;

    const int j    = threadIdx.x;
    const int lane = j & 31;
    const int wave = j >> 5;

    // Fold the 8 K-slice partials (fixed order -> deterministic).
    float uj = 0.0f, vj = 0.0f;
    #pragma unroll
    for (int s = 0; s < KSLICES; ++s) {
        uj += part[s * (2 * DIM) + j];
        vj += part[s * (2 * DIM) + DIM + j];
    }

    const float yk = y[j];
    const float t  = 1.0f / (1.0f + __expf(-uj));
    const float g  = fmaxf(vj, 0.0f);
    const float z  = t * g + (1.0f - t) * yk;

    // ---- block max ----
    float m = z;
    #pragma unroll
    for (int off = 16; off > 0; off >>= 1) m = fmaxf(m, __shfl_xor(m, off, 32));
    if (lane == 0) sred[wave] = m;
    __syncthreads();
    if (wave == 0) {
        float mm = sred[lane];
        #pragma unroll
        for (int off = 16; off > 0; off >>= 1) mm = fmaxf(mm, __shfl_xor(mm, off, 32));
        if (lane == 0) sbc = mm;
    }
    __syncthreads();
    const float M = sbc;

    // ---- block sum of exp(z - M) ----
    float s = __expf(z - M);
    #pragma unroll
    for (int off = 16; off > 0; off >>= 1) s += __shfl_xor(s, off, 32);
    if (lane == 0) sred[wave] = s;
    __syncthreads();
    if (wave == 0) {
        float ss = sred[lane];
        #pragma unroll
        for (int off = 16; off > 0; off >>= 1) ss += __shfl_xor(ss, off, 32);
        if (lane == 0) sbc = ss;
    }
    __syncthreads();

    out[j] = (z - M) - __logf(sbc);
}

// ---------------------------------------------------------------------------
// Launch: prep_y -> gemv_wmma -> finalize (same stream, graph-capture safe)
// Workspace (floats):
//   [0    .. 1023]  y (f32)
//   [1024 .. 1535]  y (f16, 1024 halves)        -- 32B-aligned at byte 4096
//   [1536 .. 17919] partials: KSLICES x 2048    -- 64 KB
// ---------------------------------------------------------------------------
extern "C" void kernel_launch(void* const* d_in, const int* in_sizes, int n_in,
                              void* d_out, int out_size, void* d_ws, size_t ws_size,
                              hipStream_t stream) {
    (void)in_sizes; (void)n_in; (void)out_size; (void)ws_size;
    // setup_inputs order: input(int,1), filters(DIM*VOCAB), w_t(DIM*DIM), w_h(DIM*DIM)
    const float* filters = (const float*)d_in[1];
    const float* w_t     = (const float*)d_in[2];
    const float* w_h     = (const float*)d_in[3];
    float* out = (float*)d_out;

    float*    ws_y    = (float*)d_ws;
    _Float16* ws_yh   = (_Float16*)(ws_y + DIM);
    float*    ws_part = ws_y + DIM + DIM / 2;

    prep_y_kernel<<<(DIM + 255) / 256, 256, 0, stream>>>(filters, ws_y, ws_yh);
    gemv_wmma_kernel<<<128, 256, 0, stream>>>(w_t, w_h, ws_yh, ws_part);
    finalize_kernel<<<1, 1024, 0, stream>>>(ws_y, ws_part, out);
}
